// GCNET_76836964925799
// MI455X (gfx1250) — compile-verified, matrix operand order checked by reading, not run on hardware
//
#include <hip/hip_runtime.h>

#define S_LEN  192
#define B_CONV 64
#define F_INN  512
#define H_GCN  128
#define D_HH   640
#define MEM_D  1280
#define GATE4  2560

typedef __attribute__((ext_vector_type(16))) __bf16 v16bf;
typedef __attribute__((ext_vector_type(8)))  float  v8f;
typedef __attribute__((ext_vector_type(4)))  unsigned int u32x4;
typedef __attribute__((ext_vector_type(8)))  int  i32x8;
typedef __attribute__((ext_vector_type(4)))  int  i32x4;

// ---------------------------------------------------------------------------
// helpers
// ---------------------------------------------------------------------------
__device__ __forceinline__ int lane_id() { return (int)(threadIdx.x & 31); }

// element e (0..15) + lane-half h -> K offset within a 32-wide K tile
// (mirrors the documented 16-bit A-matrix VGPR layout)
__device__ __forceinline__ int kmap(int e, int h) {
  return (e < 8) ? (h * 8 + e) : (16 + h * 8 + (e - 8));
}

__device__ __forceinline__ v8f zero8() {
  v8f z;
#pragma unroll
  for (int i = 0; i < 8; ++i) z[i] = 0.0f;
  return z;
}

__device__ __forceinline__ float sigm(float x) { return 1.0f / (1.0f + __expf(-x)); }

// A tile: 16x32 bf16 from row-major f32 (optional per-row scale), lane m holds row m
__device__ __forceinline__ v16bf load_a(const float* A, long sM, int M, int K,
                                        int row0, int k0, const float* rowScale) {
  int l = lane_id();
  int m = l & 15, h = l >> 4;
  int r = row0 + m;
  v16bf out;
  if (r < M) {
    const float* rp = A + (long)r * sM;
    float sc = rowScale ? rowScale[r] : 1.0f;
#pragma unroll
    for (int e = 0; e < 16; ++e) {
      int k = k0 + kmap(e, h);
      float v = (k < K) ? rp[k] * sc : 0.0f;
      out[e] = (__bf16)v;
    }
  } else {
#pragma unroll
    for (int e = 0; e < 16; ++e) out[e] = (__bf16)0.0f;
  }
  return out;
}

// B tile: 32x16 bf16 gathered from f32 with arbitrary strides (lane n holds col n)
__device__ __forceinline__ v16bf load_b_f32(const float* Bp, long sK, long sN,
                                            int K, int N, int k0, int n0) {
  int l = lane_id();
  int n = l & 15, h = l >> 4;
  int c = n0 + n;
  v16bf out;
#pragma unroll
  for (int e = 0; e < 16; ++e) {
    int k = k0 + kmap(e, h);
    float v = (k < K && c < N) ? Bp[(long)k * sK + (long)c * sN] : 0.0f;
    out[e] = (__bf16)v;
  }
  return out;
}

// B tile from pre-packed layout: contiguous 32 bytes per lane per tile
__device__ __forceinline__ v16bf load_b_packed(const __bf16* P, int ktile, int ntile,
                                               int ntiles) {
  const __bf16* p = P + (((long)ktile * ntiles + ntile) * 32 + lane_id()) * 16;
  return *(const v16bf*)p;
}

// C/D 16x16 f32 store: VGPR v -> row (lane>>4)*8+v, col lane&15
__device__ __forceinline__ void store_c(float* C, long sM, int M, int N,
                                        int row0, int n0, const v8f acc,
                                        const float* bias, int relu) {
  int l = lane_id();
  int col = n0 + (l & 15);
  if (col >= N) return;
  int rbase = row0 + (l >> 4) * 8;
  float bv = bias ? bias[col] : 0.0f;
#pragma unroll
  for (int v = 0; v < 8; ++v) {
    int r = rbase + v;
    if (r < M) {
      float x = acc[v] + bv;
      if (relu) x = fmaxf(x, 0.0f);
      C[(long)r * sM + col] = x;
    }
  }
}

#define WMMA_BF16(A, B, C) \
  __builtin_amdgcn_wmma_f32_16x16x32_bf16(false, (A), false, (B), (short)0, (C), false, false)

// ---------------------------------------------------------------------------
// utility kernels
// ---------------------------------------------------------------------------
__global__ void zero_f32(float* p, long n) {
  for (long i = blockIdx.x * (long)blockDim.x + threadIdx.x; i < n;
       i += (long)gridDim.x * blockDim.x)
    p[i] = 0.0f;
}

__global__ void vec_add(const float* a, const float* b, float* o, int n) {
  for (int i = blockIdx.x * blockDim.x + threadIdx.x; i < n; i += gridDim.x * blockDim.x)
    o[i] = a[i] + b[i];
}

__global__ void cnt_inv(const float* cnt, float* inv, long n) {
  for (long i = blockIdx.x * (long)blockDim.x + threadIdx.x; i < n;
       i += (long)gridDim.x * blockDim.x)
    inv[i] = 1.0f / fmaxf(cnt[i], 1.0f);
}

// pack W (K x N, element(k,n)=src[k*sK+n*sN]) into per-lane WMMA B-tile layout
__global__ void pack_b_bf16(const float* src, __bf16* dst, int K, int N, long sK, long sN) {
  int ktiles = (K + 31) >> 5, ntiles = (N + 15) >> 4;
  long total = (long)ktiles * ntiles * 32;
  for (long idx = blockIdx.x * (long)blockDim.x + threadIdx.x; idx < total;
       idx += (long)gridDim.x * blockDim.x) {
    int lane = (int)(idx & 31);
    long tile = idx >> 5;
    int ntile = (int)(tile % ntiles), ktile = (int)(tile / ntiles);
    int n = ntile * 16 + (lane & 15), h = lane >> 4;
    __bf16* out = dst + idx * 16;
#pragma unroll
    for (int e = 0; e < 16; ++e) {
      int k = ktile * 32 + kmap(e, h);
      float v = (k < K && n < N) ? src[(long)k * sK + (long)n * sN] : 0.0f;
      out[e] = (__bf16)v;
    }
  }
}

// ---------------------------------------------------------------------------
// graph stage
// ---------------------------------------------------------------------------
__global__ void rgcn_scatter(const float* feat, const int* ei, const int* et,
                             float* agg, float* cnt, int E, int Nn) {
  int e = blockIdx.x;
  if (e >= E) return;
  int s = ei[e], d = ei[E + e], r = et[e];
  const float* sp = feat + (long)s * F_INN;
  float* dp = agg + ((long)r * Nn + d) * F_INN;
  for (int f = threadIdx.x; f < F_INN; f += blockDim.x) atomicAdd(&dp[f], sp[f]);
  if (threadIdx.x == 0) atomicAdd(&cnt[(long)r * Nn + d], 1.0f);
}

__global__ __launch_bounds__(256)
void rgcn_gemm(const float* feat, const float* agg, const float* cinv,
               const __bf16* WrootP, const __bf16* WrelP, const float* b1,
               float* out, int Nn) {
  int mtiles = (Nn + 15) >> 4;
  int wave = (int)((blockIdx.x * (long)blockDim.x + threadIdx.x) >> 5);
  if (wave >= mtiles * 8) return;
  int mt = wave >> 3, nt = wave & 7;
  v8f acc = zero8();
  for (int kt = 0; kt < 16; ++kt) {
    v16bf a = load_a(feat, F_INN, Nn, F_INN, mt * 16, kt * 32, nullptr);
    v16bf b = load_b_packed(WrootP, kt, nt, 8);
    acc = WMMA_BF16(a, b, acc);
  }
  for (int r = 0; r < 3; ++r) {
    const float* Ar = agg + (long)r * Nn * F_INN;
    const float* sc = cinv + (long)r * Nn;
    const __bf16* Bp = WrelP + (long)r * 16 * 8 * 512;
    for (int kt = 0; kt < 16; ++kt) {
      v16bf a = load_a(Ar, F_INN, Nn, F_INN, mt * 16, kt * 32, sc);
      v16bf b = load_b_packed(Bp, kt, nt, 8);
      acc = WMMA_BF16(a, b, acc);
    }
  }
  store_c(out, H_GCN, Nn, H_GCN, mt * 16, nt * 16, acc, b1, 0);
}

__global__ void gconv_scatter(const float* out1, const int* ei, float* agg2, int E) {
  int e = blockIdx.x;
  if (e >= E) return;
  int s = ei[e], d = ei[E + e];
  const float* sp = out1 + (long)s * H_GCN;
  float* dp = agg2 + (long)d * H_GCN;
  for (int f = threadIdx.x; f < H_GCN; f += blockDim.x) atomicAdd(&dp[f], sp[f]);
}

__global__ __launch_bounds__(256)
void gconv_gemm(const float* agg2, const float* out1, const __bf16* Wrel2P,
                const __bf16* Wroot2P, const float* b2, float* h, int Nn) {
  int mtiles = (Nn + 15) >> 4;
  int wave = (int)((blockIdx.x * (long)blockDim.x + threadIdx.x) >> 5);
  if (wave >= mtiles * 8) return;
  int mt = wave >> 3, nt = wave & 7;
  v8f acc = zero8();
  for (int kt = 0; kt < 4; ++kt) {
    v16bf a = load_a(agg2, H_GCN, Nn, H_GCN, mt * 16, kt * 32, nullptr);
    v16bf b = load_b_packed(Wrel2P, kt, nt, 8);
    acc = WMMA_BF16(a, b, acc);
    v16bf a2 = load_a(out1, H_GCN, Nn, H_GCN, mt * 16, kt * 32, nullptr);
    v16bf b2t = load_b_packed(Wroot2P, kt, nt, 8);
    acc = WMMA_BF16(a2, b2t, acc);
  }
  store_c(h, H_GCN, Nn, H_GCN, mt * 16, nt * 16, acc, b2, 0);
}

__global__ void scatter_nodes(const float* feat, const float* hg, const int* ci,
                              float* x0, int Nn) {
  int i = blockIdx.x;
  if (i >= Nn) return;
  int c = ci[i];
  int b = c / S_LEN, j = c % S_LEN;
  float* dst = x0 + ((long)j * B_CONV + b) * D_HH;
  for (int f = threadIdx.x; f < D_HH; f += blockDim.x)
    dst[f] = (f < F_INN) ? feat[(long)i * F_INN + f] : hg[(long)i * H_GCN + (f - F_INN)];
}

// ---------------------------------------------------------------------------
// generic GEMM: C[MxN] = A(f32) * Bpacked(bf16) + bias, optional relu
// ---------------------------------------------------------------------------
__global__ __launch_bounds__(256)
void gemm_apk(const float* A, long sAm, const __bf16* Bp, const float* bias,
              float* C, long sCm, int M, int N, int K, int relu) {
  int ntiles = (N + 15) >> 4;
  int mtiles = (M + 15) >> 4;
  int ktiles = (K + 31) >> 5;
  int wave = (int)((blockIdx.x * (long)blockDim.x + threadIdx.x) >> 5);
  if (wave >= mtiles * ntiles) return;
  int mt = wave / ntiles, nt = wave % ntiles;
  v8f acc = zero8();
  for (int kt = 0; kt < ktiles; ++kt) {
    if (kt + 1 < ktiles)
      __builtin_prefetch(Bp + (((long)(kt + 1) * ntiles + nt) * 32 + lane_id()) * 16, 0, 0);
    v16bf a = load_a(A, sAm, M, K, mt * 16, kt * 32, nullptr);
    v16bf b = load_b_packed(Bp, kt, nt, ntiles);
    acc = WMMA_BF16(a, b, acc);
  }
  store_c(C, sCm, M, N, mt * 16, nt * 16, acc, bias, relu);
}

// batched GEMM, both operands f32 (per-conversation attention GEMMs)
__global__ __launch_bounds__(256)
void gemm_batched(const float* A, long aB, long sAm, const float* Bsrc, long bB,
                  long sBk, long sBn, float* C, long cB, long sCm, int M, int N, int K) {
  int batch = blockIdx.y;
  const float* Ab = A + aB * batch;
  const float* Bb = Bsrc + bB * batch;
  float* Cb = C + cB * batch;
  int ntiles = (N + 15) >> 4;
  int mtiles = (M + 15) >> 4;
  int ktiles = (K + 31) >> 5;
  int wave = (int)((blockIdx.x * (long)blockDim.x + threadIdx.x) >> 5);
  if (wave >= mtiles * ntiles) return;
  int mt = wave / ntiles, nt = wave % ntiles;
  v8f acc = zero8();
  for (int kt = 0; kt < ktiles; ++kt) {
    v16bf a = load_a(Ab, sAm, M, K, mt * 16, kt * 32, nullptr);
    v16bf b = load_b_f32(Bb, sBk, sBn, K, N, kt * 32, nt * 16);
    acc = WMMA_BF16(a, b, acc);
  }
  store_c(Cb, sCm, M, N, mt * 16, nt * 16, acc, nullptr, 0);
}

// ---------------------------------------------------------------------------
// LSTM recurrent step: gates = G[t] + h_prev @ Whh^T ; fused nonlinearity
// grid.x = 40 hidden tiles, 4 waves = 4 gate chunks, 4 M-tiles each (batch 64)
// Whh working set (4 gate ntiles x 20 ktiles x 1KB) staged into LDS by the
// Tensor Data Mover with a single 3-D descriptor; waves read B tiles via DS.
// dynamic LDS: [0,80KB) staged weights, [80KB,96KB) gate exchange
// ---------------------------------------------------------------------------
__global__ __launch_bounds__(128)
void lstm_step(const float* G, const float* hprev, const float* cprev,
               const __bf16* WhhP, float* hnew, float* cnew, float* outSeq,
               int t, int dirOff) {
  extern __shared__ unsigned char smem[];
  const __bf16* wlds = (const __bf16*)smem;            // [kt][gate][1KB]
  float* gl = (float*)(smem + 80 * 1024);              // [4][64][16]
  int wv = (int)(threadIdx.x >> 5);  // gate 0..3 (i,f,g,o)
  int jt = blockIdx.x;               // hidden tile 0..39
  const float* Gt = G + (long)t * B_CONV * GATE4;

  // ---- TDM: stage Whh slice for this block into LDS (wave 0 issues) ----
  if (threadIdx.x < 32) {
    unsigned long long gaddr =
        (unsigned long long)(size_t)WhhP + (unsigned long long)jt * 1024ull;
    unsigned lds_base = (unsigned)(size_t)smem;
    u32x4 g0;
    g0[0] = 1u;                                        // count=1 valid user D#
    g0[1] = lds_base;                                  // lds_addr
    g0[2] = (unsigned)(gaddr & 0xFFFFFFFFull);         // global_addr[31:0]
    g0[3] = (unsigned)((gaddr >> 32) & 0x1FFFFFFull) | (2u << 30);  // [56:32]|type=2
    i32x8 g1;
    unsigned td0 = 256u;        // tensor_dim0 in 4B units (one packed chunk)
    unsigned td1 = 4u;          // tensor_dim1 (gates)
    unsigned st0 = 10240u;      // dim0 stride: 40KB / 4 (gate stride)
    unsigned st1 = 40960u;      // dim1 stride: 160KB / 4 (ktile stride)
    g1[0] = (int)(2u << 16);                           // data_size=4B
    g1[1] = (int)((td0 & 0xFFFFu) << 16);              // td0 low16 (abar addr=0)
    g1[2] = (int)(((td0 >> 16) & 0xFFFFu) | ((td1 & 0xFFFFu) << 16));
    g1[3] = (int)(((td1 >> 16) & 0xFFFFu) | (256u << 16));  // tile_dim0=256
    g1[4] = (int)(4u | (20u << 16));                   // tile_dim1=4, tile_dim2=20
    g1[5] = (int)st0;                                  // dim0_stride[31:0]
    g1[6] = (int)((st1 & 0xFFFFu) << 16);              // d0s[47:32]=0 | d1s[15:0]
    g1[7] = (int)(st1 >> 16);                          // dim1_stride[47:16]
    i32x4 g2;
    g2[0] = 20;                                        // tensor_dim2
    g2[1] = 0;
    g2[2] = 0;
    g2[3] = 0;
    i32x4 g3;
    g3[0] = 0; g3[1] = 0; g3[2] = 0; g3[3] = 0;
    i32x8 g4;
#pragma unroll
    for (int i = 0; i < 8; ++i) g4[i] = 0;
    __builtin_amdgcn_tensor_load_to_lds(g0, g1, g2, g3, g4, 0);
    __builtin_amdgcn_s_wait_tensorcnt(0);
  }
  __syncthreads();

  v8f accs[4];
#pragma unroll
  for (int i = 0; i < 4; ++i) accs[i] = zero8();
  for (int kt = 0; kt < 20; ++kt) {
    // B tile from LDS: chunk (kt, gate wv), 32B per lane
    const __bf16* bp = wlds + ((kt * 4 + wv) * 512 + lane_id() * 16);
    v16bf b = *(const v16bf*)bp;
#pragma unroll
    for (int mtile = 0; mtile < 4; ++mtile) {
      v16bf a = load_a(hprev, D_HH, 64, D_HH, mtile * 16, kt * 32, nullptr);
      accs[mtile] = WMMA_BF16(a, b, accs[mtile]);
    }
  }
  int l = lane_id();
  int col = l & 15, rb = (l >> 4) * 8;
#pragma unroll
  for (int mtile = 0; mtile < 4; ++mtile) {
#pragma unroll
    for (int v = 0; v < 8; ++v) {
      int r = mtile * 16 + rb + v;
      gl[(wv * 64 + r) * 16 + col] =
          accs[mtile][v] + Gt[(long)r * GATE4 + wv * D_HH + jt * 16 + col];
    }
  }
  __syncthreads();
  for (int idx = threadIdx.x; idx < 64 * 16; idx += 128) {
    int r = idx >> 4, c = idx & 15;
    int hc = jt * 16 + c;
    float ig = gl[(0 * 64 + r) * 16 + c], fg = gl[(1 * 64 + r) * 16 + c];
    float gg = gl[(2 * 64 + r) * 16 + c], og = gl[(3 * 64 + r) * 16 + c];
    float cp = cprev[(long)r * D_HH + hc];
    float cn = sigm(fg) * cp + sigm(ig) * tanhf(gg);
    float hn = sigm(og) * tanhf(cn);
    cnew[(long)r * D_HH + hc] = cn;
    hnew[(long)r * D_HH + hc] = hn;
    outSeq[(long)t * B_CONV * MEM_D + (long)r * MEM_D + dirOff + hc] = hn;
  }
}

// ---------------------------------------------------------------------------
// fused masked softmax: alpha_s = m_s*exp(tanh(raw_s*m_s)) / sum
// ---------------------------------------------------------------------------
__global__ __launch_bounds__(256)
void attn_softmax(float* scores, const float* umask) {
  int row = (int)((blockIdx.x * (long)blockDim.x + threadIdx.x) >> 5);
  if (row >= S_LEN * B_CONV) return;
  int t = row / B_CONV, b = row % B_CONV;
  float* sp = scores + (long)t * B_CONV * S_LEN + (long)b * S_LEN;
  const float* mk = umask + (long)b * S_LEN;
  int l = lane_id();
  float e[6];
  float sum = 0.0f;
#pragma unroll
  for (int i = 0; i < 6; ++i) {
    int s = l * 6 + i;
    float m = mk[s];
    float z = tanhf(sp[s] * m);
    e[i] = m * __expf(z);
    sum += e[i];
  }
#pragma unroll
  for (int off = 16; off > 0; off >>= 1) sum += __shfl_xor(sum, off, 32);
  float inv = 1.0f / sum;
#pragma unroll
  for (int i = 0; i < 6; ++i) sp[l * 6 + i] = e[i] * inv;
}

// ---------------------------------------------------------------------------
// host
// ---------------------------------------------------------------------------
static inline dim3 gemm_grid(long M, long N) {
  long waves = ((M + 15) / 16) * ((N + 15) / 16);
  return dim3((unsigned)((waves * 32 + 255) / 256));
}

extern "C" void kernel_launch(void* const* d_in, const int* in_sizes, int n_in,
                              void* d_out, int out_size, void* d_ws, size_t ws_size,
                              hipStream_t stream) {
  const float* feat   = (const float*)d_in[0];
  const int*   ei     = (const int*)d_in[1];
  const int*   et     = (const int*)d_in[2];
  const int*   ci     = (const int*)d_in[3];
  const float* umask  = (const float*)d_in[4];
  const float* W_rel  = (const float*)d_in[5];
  const float* W_root = (const float*)d_in[6];
  const float* b1     = (const float*)d_in[7];
  const float* W_rel2 = (const float*)d_in[8];
  const float* b2     = (const float*)d_in[9];
  const float* W_root2= (const float*)d_in[10];
  const float *Wih[4], *Whh[4], *bih[4], *bhh[4];
  for (int i = 0; i < 4; ++i) {
    Wih[i] = (const float*)d_in[11 + 4 * i + 0];
    Whh[i] = (const float*)d_in[11 + 4 * i + 1];
    bih[i] = (const float*)d_in[11 + 4 * i + 2];
    bhh[i] = (const float*)d_in[11 + 4 * i + 3];
  }
  const float* Wm = (const float*)d_in[27];
  const float* bm = (const float*)d_in[28];
  const float* Wl = (const float*)d_in[29];
  const float* bl = (const float*)d_in[30];
  float* outp = (float*)d_out;

  const int Nn = in_sizes[0] / F_INN;
  const int E  = in_sizes[2];
  const long SB = (long)S_LEN * B_CONV;

  // workspace carve
  char* base = (char*)d_ws;
  size_t off = 0;
  auto carve = [&](size_t bytes) -> void* {
    void* p = base + off;
    off = (off + bytes + 255) & ~(size_t)255;
    return p;
  };
  float* agg    = (float*)carve((size_t)3 * Nn * F_INN * 4);
  float* cnt    = (float*)carve((size_t)3 * Nn * 4);
  float* cinv   = (float*)carve((size_t)3 * Nn * 4);
  float* out1   = (float*)carve((size_t)Nn * H_GCN * 4);
  float* agg2   = (float*)carve((size_t)Nn * H_GCN * 4);
  float* hg     = (float*)carve((size_t)Nn * H_GCN * 4);
  float* x0     = (float*)carve((size_t)SB * D_HH * 4);
  float* x1     = (float*)carve((size_t)SB * MEM_D * 4);
  float* x2     = (float*)carve((size_t)SB * MEM_D * 4);
  float* G      = (float*)carve((size_t)SB * GATE4 * 4);
  float* ha     = (float*)carve((size_t)B_CONV * D_HH * 4);
  float* hb     = (float*)carve((size_t)B_CONV * D_HH * 4);
  float* ca     = (float*)carve((size_t)B_CONV * D_HH * 4);
  float* cb     = (float*)carve((size_t)B_CONV * D_HH * 4);
  float* Xq     = (float*)carve((size_t)SB * MEM_D * 4);
  float* scores = (float*)carve((size_t)SB * S_LEN * 4);
  float* att    = (float*)carve((size_t)SB * MEM_D * 4);
  __bf16* WrootP  = (__bf16*)carve((size_t)16 * 8 * 512 * 2);
  __bf16* WrelP   = (__bf16*)carve((size_t)3 * 16 * 8 * 512 * 2);
  __bf16* Wrel2P  = (__bf16*)carve((size_t)4 * 8 * 512 * 2);
  __bf16* Wroot2P = (__bf16*)carve((size_t)4 * 8 * 512 * 2);
  __bf16* WihP[4];
  __bf16* WhhP[4];
  float*  bsum[4];
  for (int i = 0; i < 4; ++i) {
    int ktv = (i < 2) ? 20 : 40;
    WihP[i] = (__bf16*)carve((size_t)ktv * 160 * 512 * 2);
    WhhP[i] = (__bf16*)carve((size_t)20 * 160 * 512 * 2);
    bsum[i] = (float*)carve((size_t)GATE4 * 4);
  }
  __bf16* WmP = (__bf16*)carve((size_t)40 * 80 * 512 * 2);
  __bf16* WlP = (__bf16*)carve((size_t)40 * 40 * 512 * 2);
  (void)ws_size; (void)n_in; (void)out_size;

  // ---- weight packing (bf16 WMMA B-tile layout) ----
  auto packN = [&](const float* src, __bf16* dst, int K, int N, long sK, long sN) {
    int ktiles = (K + 31) >> 5, ntiles = (N + 15) >> 4;
    long total = (long)ktiles * ntiles * 32;
    int blocks = (int)((total + 255) / 256);
    pack_b_bf16<<<blocks, 256, 0, stream>>>(src, dst, K, N, sK, sN);
  };
  packN(W_root, WrootP, F_INN, H_GCN, H_GCN, 1);
  for (int r = 0; r < 3; ++r)
    packN(W_rel + (long)r * F_INN * H_GCN, WrelP + (long)r * 16 * 8 * 512,
          F_INN, H_GCN, H_GCN, 1);
  packN(W_rel2, Wrel2P, H_GCN, H_GCN, H_GCN, 1);
  packN(W_root2, Wroot2P, H_GCN, H_GCN, H_GCN, 1);
  for (int i = 0; i < 4; ++i) {
    int din = (i < 2) ? D_HH : MEM_D;
    packN(Wih[i], WihP[i], din, GATE4, 1, din);   // Wih^T: element(k,n)=Wih[n*din+k]
    packN(Whh[i], WhhP[i], D_HH, GATE4, 1, D_HH);
    vec_add<<<10, 256, 0, stream>>>(bih[i], bhh[i], bsum[i], GATE4);
  }
  packN(Wm, WmP, MEM_D, MEM_D, MEM_D, 1);
  packN(Wl, WlP, MEM_D, D_HH, D_HH, 1);

  // ---- graph stage ----
  zero_f32<<<4096, 256, 0, stream>>>(agg, (long)3 * Nn * F_INN);
  zero_f32<<<256, 256, 0, stream>>>(cnt, (long)3 * Nn);
  zero_f32<<<1024, 256, 0, stream>>>(agg2, (long)Nn * H_GCN);
  zero_f32<<<4096, 256, 0, stream>>>(x0, SB * D_HH);
  rgcn_scatter<<<E, 128, 0, stream>>>(feat, ei, et, agg, cnt, E, Nn);
  cnt_inv<<<256, 256, 0, stream>>>(cnt, cinv, (long)3 * Nn);
  {
    long waves = (long)((Nn + 15) / 16) * 8;
    rgcn_gemm<<<(unsigned)((waves * 32 + 255) / 256), 256, 0, stream>>>(
        feat, agg, cinv, WrootP, WrelP, b1, out1, Nn);
  }
  gconv_scatter<<<E, 128, 0, stream>>>(out1, ei, agg2, E);
  {
    long waves = (long)((Nn + 15) / 16) * 8;
    gconv_gemm<<<(unsigned)((waves * 32 + 255) / 256), 256, 0, stream>>>(
        agg2, out1, Wrel2P, Wroot2P, b2, hg, Nn);
  }
  scatter_nodes<<<Nn, 128, 0, stream>>>(feat, hg, ci, x0, Nn);

  // ---- 2-layer bidirectional LSTM ----
  const float* xin = x0;
  float* xout = x1;
  int din = D_HH;
  const size_t lstm_lds = 96 * 1024;
  for (int layer = 0; layer < 2; ++layer) {
    for (int dir = 0; dir < 2; ++dir) {
      int pi = layer * 2 + dir;
      // hoisted input GEMM: G = X @ Wih^T + (bih + bhh)
      gemm_apk<<<gemm_grid(SB, GATE4), 256, 0, stream>>>(
          xin, din, WihP[pi], bsum[pi], G, GATE4, (int)SB, GATE4, din, 0);
      zero_f32<<<64, 256, 0, stream>>>(ha, (long)B_CONV * D_HH);
      zero_f32<<<64, 256, 0, stream>>>(ca, (long)B_CONV * D_HH);
      float *hp = ha, *hn = hb, *cp = ca, *cn = cb;
      for (int st = 0; st < S_LEN; ++st) {
        int t = dir ? (S_LEN - 1 - st) : st;
        lstm_step<<<40, 128, lstm_lds, stream>>>(G, hp, cp, WhhP[pi], hn, cn, xout,
                                                 t, dir * D_HH);
        float* tmp;
        tmp = hp; hp = hn; hn = tmp;
        tmp = cp; cp = cn; cn = tmp;
      }
    }
    xin = x1; xout = x2; din = MEM_D;
  }
  // x2 == M (S,B,MEM)

  // ---- matching attention ----
  gemm_apk<<<gemm_grid(SB, MEM_D), 256, 0, stream>>>(
      x2, MEM_D, WmP, bm, Xq, MEM_D, (int)SB, MEM_D, MEM_D, 0);
  {  // scores[t,b,s] = Xq[t,b,:] . M[s,b,:]
    long waves = 12 * 12;
    dim3 grid((unsigned)((waves * 32 + 255) / 256), B_CONV);
    gemm_batched<<<grid, 256, 0, stream>>>(
        Xq, (long)MEM_D, (long)B_CONV * MEM_D,
        x2, (long)MEM_D, 1L, (long)B_CONV * MEM_D,
        scores, (long)S_LEN, (long)B_CONV * S_LEN, S_LEN, S_LEN, MEM_D);
  }
  attn_softmax<<<(S_LEN * B_CONV * 32 + 255) / 256, 256, 0, stream>>>(scores, umask);
  {  // att[t,b,:] = sum_s alpha[t,b,s] * M[s,b,:]
    long waves = 12 * 80;
    dim3 grid((unsigned)((waves * 32 + 255) / 256), B_CONV);
    gemm_batched<<<grid, 256, 0, stream>>>(
        scores, (long)S_LEN, (long)B_CONV * S_LEN,
        x2, (long)MEM_D, (long)B_CONV * MEM_D, 1L,
        att, (long)MEM_D, (long)B_CONV * MEM_D, S_LEN, MEM_D, S_LEN);
  }
  // final projection + relu -> d_out (S,B,D_H)
  gemm_apk<<<gemm_grid(SB, D_HH), 256, 0, stream>>>(
      att, MEM_D, WlP, bl, outp, D_HH, (int)SB, D_HH, MEM_D, 1);
}
